// OptimizedMultiAttentionHead_53412213293082
// MI455X (gfx1250) — matrix operand order repaired and checked by reading, not compiled, not run on hardware
//
#include <hip/hip_runtime.h>
#include <hip/hip_bf16.h>
#include <stdint.h>

// Problem constants (B=2, T=2048, C=1024, H=16, HD=64)
#define T_SEQ 2048
#define C_DIM 1024
#define BT    4096

#define LDSTRIDE 40                 // padded LDS row stride (shorts)
#define TILEBUF  (128 * LDSTRIDE)   // shorts per tile buffer

typedef __attribute__((ext_vector_type(16))) __bf16 v16bf;
typedef __attribute__((ext_vector_type(8)))  float  v8f;
typedef __attribute__((ext_vector_type(4)))  int    v4i;

union ABFrag { v16bf v; uint4 q[2]; };

// ---- CDNA5 async global->LDS DMA (ASYNCcnt-tracked), feature-gated ----
#if defined(__has_builtin)
#  if __has_builtin(__builtin_amdgcn_global_load_async_to_lds_b128)
#    define HAVE_ASYNC_LDS 1
#  endif
#  if __has_builtin(__builtin_amdgcn_s_wait_asynccnt)
#    define HAVE_WAIT_ASYNC 1
#  endif
#endif
#ifndef HAVE_ASYNC_LDS
#  define HAVE_ASYNC_LDS 0
#endif
#ifndef HAVE_WAIT_ASYNC
#  define HAVE_WAIT_ASYNC 0
#endif

template <int N>
__device__ __forceinline__ void wait_async() {
#if HAVE_WAIT_ASYNC
  __builtin_amdgcn_s_wait_asynccnt(N);
#else
  asm volatile("s_wait_asynccnt %0" ::"i"(N) : "memory");
#endif
}

#if HAVE_ASYNC_LDS
typedef __attribute__((address_space(1))) v4i* gbl_v4i_ptr;
typedef __attribute__((address_space(3))) v4i* lds_v4i_ptr;
__device__ __forceinline__ void async_copy_b128(unsigned short* lds_dst,
                                                const unsigned short* gsrc) {
  __builtin_amdgcn_global_load_async_to_lds_b128((gbl_v4i_ptr)gsrc,
                                                 (lds_v4i_ptr)lds_dst, 0, 0);
}
#endif

__device__ __forceinline__ unsigned short f32_to_bf16(float f) {
  union { float f; uint32_t u; } x; x.f = f;
  uint32_t lsb = (x.u >> 16) & 1u;
  x.u += 0x7fffu + lsb;                 // round-to-nearest-even
  return (unsigned short)(x.u >> 16);
}

// A fragment (16x32 bf16, MxK): lane -> M=lane&15; elems 0..7 <- K=half*8+e,
// elems 8..15 <- K=16+half*8+(e-8)   (ISA 7.12.2, 16-bit A 16x32)
__device__ __forceinline__ v16bf load_A_frag(const unsigned short* base, int stride) {
  int lane = threadIdx.x & 31;
  const unsigned short* r = base + (size_t)(lane & 15) * stride + (lane >> 4) * 8;
  ABFrag u;
  u.q[0] = *(const uint4*)(r);
  u.q[1] = *(const uint4*)(r + 16);
  return u.v;
}

// B fragment (32x16 bf16, KxN) stored [N][K]: lane -> N=lane&15; elems 0..7 <- K=half*8+e,
// elems 8..15 <- K=16+half*8+(e-8)
__device__ __forceinline__ v16bf load_B_frag(const unsigned short* base, int stride) {
  int lane = threadIdx.x & 31;
  const unsigned short* r = base + (size_t)(lane & 15) * stride + (lane >> 4) * 8;
  ABFrag u;
  u.q[0] = *(const uint4*)(r);
  u.q[1] = *(const uint4*)(r + 16);
  return u.v;
}

__device__ __forceinline__ v8f wmma_bf16(v16bf a, v16bf b, v8f c) {
  return __builtin_amdgcn_wmma_f32_16x16x32_bf16(false, a, false, b, (short)0, c, false, false);
}

__device__ __forceinline__ float half_max(float v) {
#pragma unroll
  for (int off = 1; off < 16; off <<= 1) v = fmaxf(v, __shfl_xor(v, off, 32));
  return v;
}
__device__ __forceinline__ float half_sum(float v) {
#pragma unroll
  for (int off = 1; off < 16; off <<= 1) v += __shfl_xor(v, off, 32);
  return v;
}

// ---------------- conversion kernels ----------------
__global__ void cvt_f32_to_bf16_kernel(const float* __restrict__ src,
                                       unsigned short* __restrict__ dst, int n) {
  int i = blockIdx.x * blockDim.x + threadIdx.x;
  if (i < n) dst[i] = f32_to_bf16(src[i]);
}

// src: fp32 [K][N] row-major -> dst: bf16 [N][K] (transposed so B frags are contiguous)
__global__ void cvt_transpose_kernel(const float* __restrict__ src,
                                     unsigned short* __restrict__ dst, int K, int N) {
  int i = blockIdx.x * blockDim.x + threadIdx.x;
  if (i < K * N) {
    int k = i / N, n = i % N;
    dst[(size_t)n * K + k] = f32_to_bf16(src[i]);
  }
}

// ---------------- tile staging: one 128x32 bf16 tile of A and B ----------------
__device__ __forceinline__ void issue_tile_loads(const unsigned short* __restrict__ Ag,
                                                 const unsigned short* __restrict__ Bg,
                                                 int lda, int ldb, int m0, int n0, int k0,
                                                 unsigned short* As, unsigned short* Bs) {
  const int tid  = threadIdx.x;
  const int lrow = tid >> 1;
  const int lcol = (tid & 1) * 16;
  const unsigned short* ga = Ag + (size_t)(m0 + lrow) * lda + k0 + lcol;
  const unsigned short* gb = Bg + (size_t)(n0 + lrow) * ldb + k0 + lcol;
  unsigned short* la = As + lrow * LDSTRIDE + lcol;
  unsigned short* lb = Bs + lrow * LDSTRIDE + lcol;
#if HAVE_ASYNC_LDS
  async_copy_b128(la, ga);
  async_copy_b128(la + 8, ga + 8);
  async_copy_b128(lb, gb);
  async_copy_b128(lb + 8, gb + 8);
#else
  ((uint4*)la)[0] = ((const uint4*)ga)[0];
  ((uint4*)la)[1] = ((const uint4*)ga)[1];
  ((uint4*)lb)[0] = ((const uint4*)gb)[0];
  ((uint4*)lb)[1] = ((const uint4*)gb)[1];
#endif
}

// ---------------- shared GEMM mainloop (128x128 tile, K step 32, double-buffered) --------
__device__ __forceinline__ void gemm_mainloop(const unsigned short* __restrict__ Ag,
                                              const unsigned short* __restrict__ Bg,
                                              int lda, int ldb, int K, int m0, int n0,
                                              unsigned short* As, unsigned short* Bs,
                                              v8f (&acc)[2][4]) {
  const int w  = threadIdx.x >> 5;
  const int wm = (w & 3) * 32;   // 4 waves in M
  const int wn = (w >> 2) * 64;  // 2 waves in N

  issue_tile_loads(Ag, Bg, lda, ldb, m0, n0, 0, As, Bs);
  int pg = 0;
  for (int k0 = 0; k0 < K; k0 += 32, pg ^= 1) {
    const int knext = k0 + 32;
    if (knext < K) {
      // stream next tile into the other buffer while we compute on this one
      issue_tile_loads(Ag, Bg, lda, ldb, m0, n0, knext,
                       As + (pg ^ 1) * TILEBUF, Bs + (pg ^ 1) * TILEBUF);
      wait_async<4>();   // current tile done; 4 newer (next-tile) loads may remain
    } else {
      wait_async<0>();
    }
    __syncthreads();     // all waves' DMA for this tile complete

    const unsigned short* Ab = As + pg * TILEBUF;
    const unsigned short* Bb = Bs + pg * TILEBUF;
    v16bf aF[2], bF[4];
#pragma unroll
    for (int tm = 0; tm < 2; ++tm) aF[tm] = load_A_frag(Ab + (wm + tm * 16) * LDSTRIDE, LDSTRIDE);
#pragma unroll
    for (int tn = 0; tn < 4; ++tn) bF[tn] = load_B_frag(Bb + (wn + tn * 16) * LDSTRIDE, LDSTRIDE);
#pragma unroll
    for (int tm = 0; tm < 2; ++tm)
#pragma unroll
      for (int tn = 0; tn < 4; ++tn)
        acc[tm][tn] = wmma_bf16(aF[tm], bF[tn], acc[tm][tn]);
    __syncthreads();     // done reading before this buffer is refilled (next iter)
  }
}

// ---------------- GEMM1: qkv = x @ Wqkv, scatter into q/k/vT head layouts ----------------
__global__ __launch_bounds__(256) void qkv_gemm_kernel(const unsigned short* __restrict__ xb,
                                                       const unsigned short* __restrict__ wqkvT,
                                                       unsigned short* __restrict__ qb,
                                                       unsigned short* __restrict__ kb,
                                                       unsigned short* __restrict__ vb) {
  __shared__ __align__(16) unsigned short As[2 * TILEBUF];
  __shared__ __align__(16) unsigned short Bs[2 * TILEBUF];
  const int m0 = blockIdx.x * 128, n0 = blockIdx.y * 128;
  v8f acc[2][4] = {};
  gemm_mainloop(xb, wqkvT, 1024, 1024, 1024, m0, n0, As, Bs, acc);

  const int lane = threadIdx.x & 31;
  const int w = threadIdx.x >> 5;
  const int wm = (w & 3) * 32, wn = (w >> 2) * 64;
  const int cn = lane & 15, hf = lane >> 4;   // C layout: M = r + 8*hf, N = cn
#pragma unroll
  for (int tm = 0; tm < 2; ++tm) {
#pragma unroll
    for (int tn = 0; tn < 4; ++tn) {
      int gn = n0 + wn + tn * 16 + cn;
#pragma unroll
      for (int r = 0; r < 8; ++r) {
        int gm = m0 + wm + tm * 16 + hf * 8 + r;
        float val = acc[tm][tn][r];
        int b = gm >> 11, t = gm & 2047;
        if (gn < 1024) {            // q, fold HD^-0.5 = 0.125
          int h = gn >> 6, hd = gn & 63;
          qb[(((size_t)b * 16 + h) * 2048 + t) * 64 + hd] = f32_to_bf16(val * 0.125f);
        } else if (gn < 2048) {     // k
          int cc = gn - 1024, h = cc >> 6, hd = cc & 63;
          kb[(((size_t)b * 16 + h) * 2048 + t) * 64 + hd] = f32_to_bf16(val);
        } else {                    // v stored transposed [B][H][HD][T]
          int cc = gn - 2048, h = cc >> 6, hd = cc & 63;
          vb[(((size_t)b * 16 + h) * 64 + hd) * 2048 + t] = f32_to_bf16(val);
        }
      }
    }
  }
}

// ---------------- attention: two-pass causal softmax + learned cos mask + PV ----------------
__global__ __launch_bounds__(256) void attn_kernel(const unsigned short* __restrict__ qg,
                                                   const unsigned short* __restrict__ kg,
                                                   const unsigned short* __restrict__ vtg,
                                                   const float* __restrict__ A1,
                                                   const float* __restrict__ B1,
                                                   unsigned short* __restrict__ attn_out) {
  __shared__ __align__(16) unsigned short Pst[8][16 * LDSTRIDE];  // per-wave P staging
  const int qt = blockIdx.x;            // query tile of 128 (16 tiles)
  const int bh = blockIdx.y;            // b*16 + h (32)
  const int w = threadIdx.x >> 5;
  const int lane = threadIdx.x & 31;
  const int cn = lane & 15, hf = lane >> 4;
  const int q0 = qt * 128 + w * 16;     // this wave's 16 query rows

  const unsigned short* qbase = qg + ((size_t)bh * 2048 + q0) * 64;
  const unsigned short* kbase = kg + (size_t)bh * 2048 * 64;
  const unsigned short* vbase = vtg + (size_t)bh * 64 * 2048;

  v16bf qa0 = load_A_frag(qbase, 64);        // hd 0..31
  v16bf qa1 = load_A_frag(qbase + 32, 64);   // hd 32..63

  float mrun[8], zsum[8];
#pragma unroll
  for (int r = 0; r < 8; ++r) { mrun[r] = -3.0e38f; zsum[r] = 0.f; }
  const int kend = q0 + 16;  // causal: keys < kend

  // ---- pass 1: row max / row sum ----
  for (int n0 = 0; n0 < kend; n0 += 32) {
#pragma unroll
    for (int nt = 0; nt < 2; ++nt) {
      int nb = n0 + nt * 16;
      v16bf kb0 = load_B_frag(kbase + (size_t)nb * 64, 64);
      v16bf kb1 = load_B_frag(kbase + (size_t)nb * 64 + 32, 64);
      v8f s = {};
      s = wmma_bf16(qa0, kb0, s);
      s = wmma_bf16(qa1, kb1, s);
      int ng = nb + cn;
#pragma unroll
      for (int r = 0; r < 8; ++r) {
        int gm = q0 + hf * 8 + r;
        float sv = (ng <= gm) ? s[r] : -3.0e38f;
        float mx = fmaxf(mrun[r], half_max(sv));
        float es = half_sum(__expf(sv - mx));
        zsum[r] = zsum[r] * __expf(mrun[r] - mx) + es;
        mrun[r] = mx;
      }
    }
  }
  float zinv[8];
#pragma unroll
  for (int r = 0; r < 8; ++r) zinv[r] = 1.0f / zsum[r];

  // ---- pass 2: P = f(softmax) -> LDS transpose -> PV WMMA ----
  v8f O[4] = {};
  unsigned short* stage = Pst[w];
  for (int n0 = 0; n0 < kend; n0 += 32) {
#pragma unroll
    for (int nt = 0; nt < 2; ++nt) {
      int nb = n0 + nt * 16;
      v16bf kb0 = load_B_frag(kbase + (size_t)nb * 64, 64);
      v16bf kb1 = load_B_frag(kbase + (size_t)nb * 64 + 32, 64);
      v8f s = {};
      s = wmma_bf16(qa0, kb0, s);
      s = wmma_bf16(qa1, kb1, s);
      int ng = nb + cn;
      float a1 = A1[ng], b1 = B1[ng];
#pragma unroll
      for (int r = 0; r < 8; ++r) {
        int gm = q0 + hf * 8 + r;
        float p = 0.f;
        if (ng <= gm) {
          p = __expf(s[r] - mrun[r]) * zinv[r];
          p = p * (0.5f * __cosf(a1 * p + b1) + 0.5f);   // learned attn dropout
        }
        stage[(hf * 8 + r) * LDSTRIDE + nt * 16 + cn] = f32_to_bf16(p);
      }
    }
    asm volatile("s_wait_dscnt 0" ::: "memory");  // P stage RAW fence (wave-local)
    v16bf pa = load_A_frag(stage, LDSTRIDE);      // P as 16x32 A fragment
#pragma unroll
    for (int tn = 0; tn < 4; ++tn) {
      v16bf vfr = load_B_frag(vbase + (size_t)(tn * 16) * 2048 + n0, 2048);
      O[tn] = wmma_bf16(pa, vfr, O[tn]);
    }
  }

  const int b = bh >> 4, h = bh & 15;
#pragma unroll
  for (int tn = 0; tn < 4; ++tn) {
#pragma unroll
    for (int r = 0; r < 8; ++r) {
      int row = b * 2048 + q0 + hf * 8 + r;
      int col = h * 64 + tn * 16 + cn;
      attn_out[(size_t)row * 1024 + col] = f32_to_bf16(O[tn][r]);
    }
  }
}

// ---------------- GEMM2: out = attn_out @ Wproj, fused learned output mask ----------------
__global__ __launch_bounds__(256) void proj_gemm_kernel(const unsigned short* __restrict__ ab,
                                                        const unsigned short* __restrict__ wprojT,
                                                        const float* __restrict__ A2,
                                                        const float* __restrict__ B2,
                                                        float* __restrict__ out) {
  __shared__ __align__(16) unsigned short As[2 * TILEBUF];
  __shared__ __align__(16) unsigned short Bs[2 * TILEBUF];
  const int m0 = blockIdx.x * 128, n0 = blockIdx.y * 128;
  v8f acc[2][4] = {};
  gemm_mainloop(ab, wprojT, 1024, 1024, 1024, m0, n0, As, Bs, acc);

  const int lane = threadIdx.x & 31;
  const int w = threadIdx.x >> 5;
  const int wm = (w & 3) * 32, wn = (w >> 2) * 64;
  const int cn = lane & 15, hf = lane >> 4;
#pragma unroll
  for (int tm = 0; tm < 2; ++tm) {
#pragma unroll
    for (int tn = 0; tn < 4; ++tn) {
      int gn = n0 + wn + tn * 16 + cn;
      float a2 = A2[gn], b2 = B2[gn];
#pragma unroll
      for (int r = 0; r < 8; ++r) {
        int gm = m0 + wm + tm * 16 + hf * 8 + r;
        float val = acc[tm][tn][r];
        float mv = 0.5f * __cosf(a2 * val + b2) + 0.5f;  // learned output dropout
        out[(size_t)gm * 1024 + gn] = val * mv;
      }
    }
  }
}

extern "C" void kernel_launch(void* const* d_in, const int* in_sizes, int n_in,
                              void* d_out, int out_size, void* d_ws, size_t ws_size,
                              hipStream_t stream) {
  const float* x     = (const float*)d_in[0];
  const float* Wqkv  = (const float*)d_in[1];
  const float* Wproj = (const float*)d_in[2];
  const float* A1    = (const float*)d_in[3];
  const float* B1    = (const float*)d_in[4];
  const float* A2    = (const float*)d_in[5];
  const float* B2    = (const float*)d_in[6];
  float* out = (float*)d_out;

  // workspace layout (bf16 elements): 48 MB total
  unsigned short* ws     = (unsigned short*)d_ws;
  unsigned short* xb     = ws;                          // x bf16           [4096][1024]
  unsigned short* wqkvT  = xb     + (size_t)4194304;    // Wqkv^T bf16      [3072][1024]
  unsigned short* wprojT = wqkvT  + (size_t)3145728;    // Wproj^T bf16     [1024][1024]
  unsigned short* qb     = wprojT + (size_t)1048576;    // q bf16           [B][H][T][64]
  unsigned short* kb     = qb     + (size_t)4194304;    // k bf16           [B][H][T][64]
  unsigned short* vb     = kb     + (size_t)4194304;    // v^T bf16         [B][H][64][T]
  unsigned short* attb   = vb     + (size_t)4194304;    // attn out bf16    [4096][1024]

  cvt_f32_to_bf16_kernel<<<4194304 / 256, 256, 0, stream>>>(x, xb, 4194304);
  cvt_transpose_kernel<<<(1024 * 3072) / 256, 256, 0, stream>>>(Wqkv, wqkvT, 1024, 3072);
  cvt_transpose_kernel<<<(1024 * 1024) / 256, 256, 0, stream>>>(Wproj, wprojT, 1024, 1024);

  qkv_gemm_kernel<<<dim3(32, 24), 256, 0, stream>>>(xb, wqkvT, qb, kb, vb);
  attn_kernel<<<dim3(16, 32), 256, 0, stream>>>(qb, kb, vb, A1, B1, attb);
  proj_gemm_kernel<<<dim3(32, 8), 256, 0, stream>>>(attb, wprojT, A2, B2, out);
}
